// Attention_62603443306943
// MI455X (gfx1250) — matrix operand order repaired and checked by reading, not compiled, not run on hardware
//
#include <hip/hip_runtime.h>
#include <hip/hip_bf16.h>
#include <cmath>

typedef __attribute__((ext_vector_type(16))) _Float16 v16h;
typedef __attribute__((ext_vector_type(8)))  _Float16 v8h;
typedef __attribute__((ext_vector_type(8)))  float    v8f;
typedef __attribute__((ext_vector_type(4)))  float    v4f;

#define HEADS 4
#define DHEAD 32
#define NTOK  4096     // 64*64
#define CDIM  256
#define HID   128      // HEADS*DHEAD
#define QKV_O 384      // 3*HID
#define ATT_SCALE 0.17677669529663687f  // 32^-0.5

// --- CDNA5 WMMA f16 16x16x32 fragment layouts (ISA 7.12.2) ---
// A (16x32): lane<16 -> M=lane, K halves {0..7,16..23}; lane>=16 -> M=lane-16,
//            K halves {8..15,24..31}  => two contiguous 8-half runs.
// B (32x16): lane -> column N=lane&15; this code loads K=0..15 (lanes 0-15) or
//            K=16..31 (lanes 16-31)   in A and B alike => one contiguous 16-half run; ISA B split = A's.
// C/D f32:   VGPR i holds M=i (lanes 0-15) or M=8+i (lanes 16-31), N=lane&15.

// B-fragment from f16 memory where K is innermost-contiguous.
// base points at K=0 of this lane's column row. 2x b128 loads.
__device__ __forceinline__ v16h frag_b_f16(const _Float16* base, int lane) {
  const v8h* p = (const v8h*)(base + ((lane & 16) ? 16 : 0));
  v8h lo = p[0], hi = p[1];
  v16h r;
#pragma unroll
  for (int h = 0; h < 8; ++h) { r[h] = lo[h]; r[h + 8] = hi[h]; }
  return r;
}

// A-fragment from f16 memory, row-major rows of >=32 halves. 2x b128 loads.
__device__ __forceinline__ v16h frag_a_f16(const _Float16* row, int lane) {
  int off = (lane & 16) ? 16 : 0;
  const v8h* p0 = (const v8h*)(row + off);
  const v8h* p1 = (const v8h*)(row + off + 8);
  v8h lo = p0[0], hi = p1[0];
  v16h r;
#pragma unroll
  for (int h = 0; h < 8; ++h) { r[h] = lo[h]; r[h + 8] = hi[h]; }
  return r;
}

// A-fragment from f32 memory with scalar scale folded in. 4x b128 f32 loads.
__device__ __forceinline__ v16h frag_a_f32(const float* row, int lane, float s) {
  int off = (lane & 16) ? 16 : 0;
  const v4f* p0 = (const v4f*)(row + off);
  const v4f* p1 = (const v4f*)(row + off + 8);
  v4f a0 = p0[0], a1 = p0[1], b0 = p1[0], b1 = p1[1];
  v16h r;
#pragma unroll
  for (int j = 0; j < 4; ++j) {
    r[j]      = (_Float16)(a0[j] * s);
    r[4 + j]  = (_Float16)(a1[j] * s);
    r[8 + j]  = (_Float16)(b0[j] * s);
    r[12 + j] = (_Float16)(b1[j] * s);
  }
  return r;
}

// ---------------------------------------------------------------------------
// Kernel 1: channel-L2 norm per spatial position; store 16/max(||x||,eps)
// (sqrt(256) folded in).
// ---------------------------------------------------------------------------
__global__ void k_rnorm(const float* __restrict__ x, float* __restrict__ rnorm) {
  int b = blockIdx.y;
  int n = blockIdx.x * blockDim.x + threadIdx.x;
  const float* xp = x + (size_t)b * CDIM * NTOK + n;
  float ss = 0.f;
#pragma unroll 4
  for (int c = 0; c < CDIM; ++c) {
    float v = xp[(size_t)c * NTOK];
    ss += v * v;
  }
  float nrm = fmaxf(sqrtf(ss), 1e-12f);
  rnorm[b * NTOK + n] = 16.0f / nrm;
}

// ---------------------------------------------------------------------------
// Kernel 2: QKV projection. qkv[o,n] = sum_c w_qkv[o,c] * x[c,n]*g[c]*rn[n].
// SCALE folded into q rows (o<128) and v rows (o>=256).
// Outputs: q,k as f16 [b][h][n][f] (row-major f);
//          v TRANSPOSED as f16 [b][h][f][n] so the attention O-matmul's
//          B-fragments are contiguous b128 loads.
// One wave per 16x16 output tile; K=256 -> 8 WMMAs.
// ---------------------------------------------------------------------------
__global__ void k_qkv(const float* __restrict__ x, const float* __restrict__ g,
                      const float* __restrict__ w_qkv, const float* __restrict__ rnorm,
                      _Float16* __restrict__ qb, _Float16* __restrict__ kb,
                      _Float16* __restrict__ vt) {
  int lane = threadIdx.x & 31;
  int wave = threadIdx.x >> 5;
  int b  = blockIdx.z;
  int o0 = blockIdx.y * 16;
  int n0 = (blockIdx.x * 4 + wave) * 16;
  int col = lane & 15;
  int nq  = n0 + col;

  float rn = rnorm[b * NTOK + nq];
  float srow = (o0 < HID || o0 >= 2 * HID) ? ATT_SCALE : 1.0f;  // q & v scaled
  const float* xb = x + (size_t)b * CDIM * NTOK;
  const float* wrow = w_qkv + (size_t)(o0 + col) * CDIM;

  v8f acc = {};
  for (int kc = 0; kc < CDIM; kc += 32) {
    v16h a = frag_a_f32(wrow + kc, lane, srow);

    // g for this lane's 16 contiguous channels (vector loads)
    int c0 = kc + ((lane & 16) ? 16 : 0);
    const v4f* gp = (const v4f*)(g + c0);
    v4f g0 = gp[0], g1 = gp[1], g2 = gp[2], g3 = gp[3];
    float gg[16];
#pragma unroll
    for (int j = 0; j < 4; ++j) {
      gg[j] = g0[j]; gg[4 + j] = g1[j]; gg[8 + j] = g2[j]; gg[12 + j] = g3[j];
    }

    // B fragment: x column gather (channel-strided input layout)
    v16h bm;
#pragma unroll
    for (int h = 0; h < 16; ++h)
      bm[h] = (_Float16)(xb[(size_t)(c0 + h) * NTOK + nq] * gg[h] * rn);

    acc = __builtin_amdgcn_wmma_f32_16x16x32_f16(false, a, false, bm,
                                                 (short)0, acc, false, false);
  }

  int head = (o0 & 127) >> 5;  // uniform per block
  int rowb = (lane & 16) ? 8 : 0;
  if (o0 < 2 * HID) {          // q and k: [b][h][n][f] (contiguous f writes)
    _Float16* dst = (o0 < HID) ? qb : kb;
    size_t rowbase = ((size_t)(b * HEADS + head) * NTOK + nq) * DHEAD;
#pragma unroll
    for (int i = 0; i < 8; ++i)
      dst[rowbase + ((o0 + rowb + i) & 31)] = (_Float16)acc[i];
  } else {                     // v: transposed [b][h][f][n]
    size_t base = (size_t)(b * HEADS + head) * DHEAD * NTOK;
#pragma unroll
    for (int i = 0; i < 8; ++i)
      vt[base + (size_t)((o0 + rowb + i) & 31) * NTOK + nq] = (_Float16)acc[i];
  }
}

// ---------------------------------------------------------------------------
// Kernel 3: flash attention. One wave per 16-row query tile, streams 32 keys
// per iteration: 2 WMMAs for S, online softmax in f32 through LDS (which also
// performs the C->A layout transpose of P), 2 WMMAs for the O accumulator.
// All global fragment traffic is b128; P readback is ds_load_b128.
// Output written as f16 att[b][n][c] (c = h*32+f) so k_out's B-fragments are
// contiguous.
// ---------------------------------------------------------------------------
__global__ void k_attn(const _Float16* __restrict__ qb, const _Float16* __restrict__ kb,
                       const _Float16* __restrict__ vt, _Float16* __restrict__ att) {
  int lane = threadIdx.x & 31;
  int wave = threadIdx.x >> 5;
  int bh = blockIdx.y;                 // batch*HEADS + head
  int n0 = (blockIdx.x * 4 + wave) * 16;
  int col = lane & 15;
  int rowb = (lane & 16) ? 8 : 0;

  __shared__ __align__(16) float    Ssh[4][16][32];
  __shared__ __align__(16) _Float16 Psh[4][16][32];
  __shared__ float Msh[4][16], Lsh[4][16], Csh[4][16];

  const _Float16* Q  = qb + (size_t)bh * NTOK * DHEAD;
  const _Float16* Kp = kb + (size_t)bh * NTOK * DHEAD;
  const _Float16* Vt = vt + (size_t)bh * DHEAD * NTOK;

  // Q A-fragment (M=query row, K=feature): 2x b128, loaded once
  v16h aq = frag_a_f16(Q + (size_t)(n0 + col) * DHEAD, lane);

  if (lane < 16) { Msh[wave][lane] = -INFINITY; Lsh[wave][lane] = 0.f; }
  v8f o0a = {}, o1a = {};
  __syncthreads();

  for (int kc = 0; kc < NTOK; kc += 32) {
    // S = Q K^T over 32 keys: B columns = key rows, K halves = features
    v16h bk0 = frag_b_f16(Kp + (size_t)(kc + col) * DHEAD, lane);
    v16h bk1 = frag_b_f16(Kp + (size_t)(kc + 16 + col) * DHEAD, lane);
    v8f s0 = {}, s1 = {};
    s0 = __builtin_amdgcn_wmma_f32_16x16x32_f16(false, aq, false, bk0, (short)0, s0, false, false);
    s1 = __builtin_amdgcn_wmma_f32_16x16x32_f16(false, aq, false, bk1, (short)0, s1, false, false);

#pragma unroll
    for (int i = 0; i < 8; ++i) {
      Ssh[wave][rowb + i][col]      = s0[i];
      Ssh[wave][rowb + i][16 + col] = s1[i];
    }
    __syncthreads();

    if (lane < 16) {  // one lane per query row: online softmax update
      float mOld = Msh[wave][lane];
      float mNew = mOld;
#pragma unroll
      for (int j = 0; j < 32; ++j) mNew = fmaxf(mNew, Ssh[wave][lane][j]);
      float corr = __expf(mOld - mNew);
      float lsum = 0.f;
#pragma unroll
      for (int j = 0; j < 32; ++j) {
        float p = __expf(Ssh[wave][lane][j] - mNew);
        Psh[wave][lane][j] = (_Float16)p;
        lsum += p;
      }
      Msh[wave][lane] = mNew;
      Lsh[wave][lane] = Lsh[wave][lane] * corr + lsum;
      Csh[wave][lane] = corr;
    }
    __syncthreads();

    // rescale O accumulators by per-row correction
#pragma unroll
    for (int i = 0; i < 8; ++i) {
      float c = Csh[wave][rowb + i];
      o0a[i] *= c;
      o1a[i] *= c;
    }

    // P A-fragment (M=query row, K=key in chunk) from LDS: 2x ds_load_b128
    v16h ap = frag_a_f16(&Psh[wave][col][0], lane);

    // V B-fragments from transposed vt: contiguous key runs, 2x b128 each
    v16h bv0 = frag_b_f16(Vt + (size_t)col * NTOK + kc, lane);
    v16h bv1 = frag_b_f16(Vt + (size_t)(16 + col) * NTOK + kc, lane);
    o0a = __builtin_amdgcn_wmma_f32_16x16x32_f16(false, ap, false, bv0, (short)0, o0a, false, false);
    o1a = __builtin_amdgcn_wmma_f32_16x16x32_f16(false, ap, false, bv1, (short)0, o1a, false, false);
    __syncthreads();
  }

  // normalize by row sum; write f16 att[b][n][h*32+f]
  int b  = bh >> 2;
  int hh = bh & 3;
#pragma unroll
  for (int i = 0; i < 8; ++i) {
    float invl = 1.0f / Lsh[wave][rowb + i];
    size_t nrow = ((size_t)b * NTOK + (n0 + rowb + i)) * HID + hh * DHEAD;
    att[nrow + col]      = (_Float16)(o0a[i] * invl);
    att[nrow + 16 + col] = (_Float16)(o1a[i] * invl);
  }
}

// ---------------------------------------------------------------------------
// Kernel 4: output projection out[o,n] = sum_c w_out[o,c]*att[n,c] + b_out[o].
// K=128 -> 4 WMMAs per 16x16 tile; both operands are b128 fragment loads.
// ---------------------------------------------------------------------------
__global__ void k_out(const _Float16* __restrict__ att, const float* __restrict__ w_out,
                      const float* __restrict__ b_out, float* __restrict__ out) {
  int lane = threadIdx.x & 31;
  int wave = threadIdx.x >> 5;
  int b  = blockIdx.z;
  int o0 = blockIdx.y * 16;
  int n0 = (blockIdx.x * 4 + wave) * 16;
  int col = lane & 15;
  int nq  = n0 + col;

  const float* wrow = w_out + (size_t)(o0 + col) * HID;
  const _Float16* arow = att + ((size_t)b * NTOK + nq) * HID;

  v8f acc = {};
#pragma unroll
  for (int kc = 0; kc < HID; kc += 32) {
    v16h a  = frag_a_f32(wrow + kc, lane, 1.0f);
    v16h bm = frag_b_f16(arow + kc, lane);
    acc = __builtin_amdgcn_wmma_f32_16x16x32_f16(false, a, false, bm,
                                                 (short)0, acc, false, false);
  }
  int rowb = (lane & 16) ? 8 : 0;
#pragma unroll
  for (int i = 0; i < 8; ++i) {
    int o = o0 + rowb + i;
    out[((size_t)b * CDIM + o) * NTOK + nq] = acc[i] + b_out[o];
  }
}

// ---------------------------------------------------------------------------
extern "C" void kernel_launch(void* const* d_in, const int* in_sizes, int n_in,
                              void* d_out, int out_size, void* d_ws, size_t ws_size,
                              hipStream_t stream) {
  (void)in_sizes; (void)n_in; (void)out_size; (void)ws_size;
  const float* x     = (const float*)d_in[0];
  const float* g     = (const float*)d_in[1];
  const float* w_qkv = (const float*)d_in[2];
  const float* w_out = (const float*)d_in[3];
  const float* b_out = (const float*)d_in[4];
  float* out = (float*)d_out;

  // workspace layout (~8.03 MB):
  //   rnorm f32 [2*4096]            : 32 KB
  //   qb,kb f16 [b][h][n][f]        : 2 MB each
  //   vt    f16 [b][h][f][n]        : 2 MB
  //   att   f16 [b][n][c]           : 2 MB
  char* ws = (char*)d_ws;
  float*    rnorm = (float*)ws;
  _Float16* qb  = (_Float16*)(ws + 32768);
  _Float16* kb  = qb + (size_t)2 * HEADS * NTOK * DHEAD;
  _Float16* vt  = kb + (size_t)2 * HEADS * NTOK * DHEAD;
  _Float16* att = vt + (size_t)2 * HEADS * NTOK * DHEAD;

  // 1) channel norms: 2 batches x 4096 positions
  k_rnorm<<<dim3(NTOK / 256, 2), 256, 0, stream>>>(x, rnorm);
  // 2) QKV projection: n-tiles(256, 4 waves/block) x o-tiles(24) x batches(2)
  k_qkv<<<dim3(64, QKV_O / 16, 2), 128, 0, stream>>>(x, g, w_qkv, rnorm, qb, kb, vt);
  // 3) flash attention: q-tiles(256, 4 waves/block) x (batch*head)(8)
  k_attn<<<dim3(64, 2 * HEADS), 128, 0, stream>>>(qb, kb, vt, att);
  // 4) output projection: n-tiles x o-tiles(16) x batches(2)
  k_out<<<dim3(64, CDIM / 16, 2), 128, 0, stream>>>(att, w_out, b_out, out);
}
